// KPModel_28647431864763
// MI455X (gfx1250) — matrix operand order repaired and checked by reading, not compiled, not run on hardware
//
#include <hip/hip_runtime.h>
#include <hip/hip_bf16.h>

#define DEVINL __device__ __forceinline__

typedef unsigned short u16;
typedef __attribute__((ext_vector_type(16))) __bf16 v16bf;
typedef __attribute__((ext_vector_type(8)))  float  v8f;

constexpr int B_ = 64, P_ = 512, N_ = 512, E_ = 128, F_ = 512, L_ = 6;
constexpr float SQRT_E_INV = 1.0f / 11.313708498984761f;

// ---------- helpers ----------------------------------------------------------

DEVINL u16 cvt_bf16(float f) {            // f32 -> bf16 (native cvt if available)
  union { __bf16 h; u16 u; } x;
  x.h = (__bf16)f;
  return x.u;
}

DEVINL float fast_rcp(float x) {
#if __has_builtin(__builtin_amdgcn_rcpf)
  return __builtin_amdgcn_rcpf(x);        // v_rcp_f32, ~1ulp: plenty for bf16 data
#else
  return 1.f / x;
#endif
}

DEVINL v8f wmma_bf16(v16bf a, v16bf b, v8f c) {
  return __builtin_amdgcn_wmma_f32_16x16x32_bf16(false, a, false, b,
                                                 (short)0, c, false, false);
}

// A tile 16(M)x32(K), bf16 row-major, lda in elements.
// Layout (ISA 7.12.2): lanes 0-15 = rows, elems 0-7:K[kb..kb+7], 8-15:K[kb+16..kb+23], kb=(lane>>4)*8
DEVINL v16bf load_a_bf16(const u16* A, int lda) {
  int lane = threadIdx.x & 31;
  int m = lane & 15, kb = (lane >> 4) << 3;
  union { v16bf v; uint4 q[2]; } r;
  r.q[0] = *(const uint4*)(A + (long)m * lda + kb);
  r.q[1] = *(const uint4*)(A + (long)m * lda + kb + 16);
  return r.v;
}

// A tile = bf16(exp(scale*D [+ M])) computed on the fly from f32 source.
DEVINL v16bf load_a_exp(const float* D, const float* M, int ld, float scale) {
  int lane = threadIdx.x & 31;
  int m = lane & 15, kb = (lane >> 4) << 3;
  const float* p = D + (long)m * ld + kb;
  const float* q = M ? (M + (long)m * ld + kb) : nullptr;
  union { v16bf v; u16 u[16]; } r;
#pragma unroll
  for (int i = 0; i < 8; ++i) {
    float a0 = scale * p[i]      + (q ? q[i]      : 0.f);
    float a1 = scale * p[16 + i] + (q ? q[16 + i] : 0.f);
    r.u[i]     = cvt_bf16(__expf(a0));
    r.u[8 + i] = cvt_bf16(__expf(a1));
  }
  return r.v;
}

// B tile 32(K)x16(N) taken from transposed storage WT row-major [Nout, K]:
// lane 0-15: col n0+lane, elems 0-7:K[kb..kb+7], 8-15:K[kb+16..kb+23], kb=k0+(lane>>4)*8.
DEVINL v16bf load_bt(const u16* WT, int ldk, int n0, int k0) {
  int lane = threadIdx.x & 31;
  int c  = n0 + (lane & 15);
  int kk = k0 + ((lane >> 4) << 3);
  const uint4* p = (const uint4*)(WT + (long)c * ldk + kk);
  union { v16bf v; uint4 q[2]; } r;
  r.q[0] = p[0]; r.q[1] = p[2];
  return r.v;
}

// ---------- elementwise / setup kernels -------------------------------------

// f32 [nmat,R,C] -> bf16 transposed [nmat,C,R]
__global__ void k_cvt_t(const float* __restrict__ src, u16* __restrict__ dst,
                        int Rr, int Cc, int nmat) {
  long idx = (long)blockIdx.x * 256 + threadIdx.x;
  long tot = (long)nmat * Rr * Cc;
  if (idx >= tot) return;
  long per = (long)Rr * Cc;
  int mat = (int)(idx / per);
  int rem = (int)(idx % per);
  int r = rem / Cc, c = rem % Cc;
  dst[(long)mat * per + (long)c * Rr + r] = cvt_bf16(src[idx]);
}

__global__ void k_embed(const float* __restrict__ data, const float* __restrict__ W,
                        const float* __restrict__ bias, float* __restrict__ X,
                        u16* __restrict__ XB) {
  int idx = blockIdx.x * 256 + threadIdx.x;   // over B*N*E
  int e  = idx & (E_ - 1);
  int bn = idx >> 7;
  float v = data[bn * 2] * W[e] + data[bn * 2 + 1] * W[E_ + e] + bias[e];
  X[idx] = v; XB[idx] = cvt_bf16(v);
}

// ---------- fused QKV (and decoder KV) GEMM ---------------------------------
// grid = B*32 blocks of 128 threads (4 waves); each wave does 2 col-tiles of E.
// Outputs: Q <- sigmoid(x@Wq) f32 [B,N,E]; KVT <- bf16 [B, 2E, N] with rows
// [0,E) = exp(k)*v and [E,2E) = exp(k), transposed for the AFT GEMM's B loads.
// The transpose goes through LDS so global stores are coalesced 32B runs.
template <bool HAS_Q>
__global__ void k_qkv(const u16* __restrict__ XB, const u16* __restrict__ WQT,
                      const u16* __restrict__ WKT, const u16* __restrict__ WVT,
                      float* __restrict__ Q, u16* __restrict__ KVT) {
  constexpr int LSTR = 20;                 // padded row stride (u16) -> bank spread
  __shared__ u16 tlds[256 * LSTR];
  int b = blockIdx.x >> 5, mt = blockIdx.x & 31;
  int n0 = mt * 16;
  int wave = threadIdx.x >> 5;
  const u16* A  = XB + (long)b * N_ * E_ + (long)n0 * E_;
  u16* kvb      = KVT + (long)b * 2 * E_ * N_;
  v8f aq[2] = {}, ak[2] = {}, av[2] = {};
  for (int kt = 0; kt < E_ / 32; ++kt) {
    int k0 = kt * 32;
    v16bf a = load_a_bf16(A + k0, E_);
#pragma unroll
    for (int j = 0; j < 2; ++j) {
      int e0 = (wave * 2 + j) * 16;
      if (HAS_Q) aq[j] = wmma_bf16(a, load_bt(WQT, E_, e0, k0), aq[j]);
      ak[j] = wmma_bf16(a, load_bt(WKT, E_, e0, k0), ak[j]);
      av[j] = wmma_bf16(a, load_bt(WVT, E_, e0, k0), av[j]);
    }
  }
  int lane = threadIdx.x & 31;
  int col = lane & 15, r0 = (lane >> 4) * 8;
#pragma unroll
  for (int j = 0; j < 2; ++j) {
    int e = (wave * 2 + j) * 16 + col;
#pragma unroll
    for (int i = 0; i < 8; ++i) {
      int rl = r0 + i;
      float ek = __expf(ak[j][i]);
      tlds[e * LSTR + rl]          = cvt_bf16(ek * av[j][i]);    // ekv rows 0..127
      tlds[(E_ + e) * LSTR + rl]   = cvt_bf16(ek);               // ek rows 128..255
      if (HAS_Q)
        Q[(long)b * N_ * E_ + (long)(n0 + rl) * E_ + e] =
            fast_rcp(1.f + __expf(-aq[j][i]));
    }
  }
  __syncthreads();
  // coalesced copy-out: thread t handles transposed rows t and t+128
  int t = threadIdx.x;
#pragma unroll
  for (int rr = 0; rr < 2; ++rr) {
    int r = t + rr * 128;                  // row in [0,2E): global row index == r
    const u16* src = tlds + r * LSTR;
    uint4* dstp = (uint4*)(kvb + (long)r * N_ + n0);
    uint4 w0, w1;
    w0.x = src[0] | ((unsigned)src[1] << 16);
    w0.y = src[2] | ((unsigned)src[3] << 16);
    w0.z = src[4] | ((unsigned)src[5] << 16);
    w0.w = src[6] | ((unsigned)src[7] << 16);
    w1.x = src[8] | ((unsigned)src[9] << 16);
    w1.y = src[10] | ((unsigned)src[11] << 16);
    w1.z = src[12] | ((unsigned)src[13] << 16);
    w1.w = src[14] | ((unsigned)src[15] << 16);
    dstp[0] = w0; dstp[1] = w1;
  }
}

// ---------- encoder AFT mixing ----------------------------------------------
// ad = exp(s*dist) computed in the A loader; num = ad@(ek*v), den = ad@ek.
// Y = X + sigmoid(q) * num/den.   grid = B*32 blocks of 64 (2 waves: E halves).
__global__ void k_aft_enc(const float* __restrict__ dist, const float* __restrict__ X,
                          const float* __restrict__ Q, const u16* __restrict__ KVT,
                          float* __restrict__ Y, const float* __restrict__ ls_p,
                          const float* __restrict__ al_p, int l) {
  int b = blockIdx.x >> 5, mt = blockIdx.x & 31;
  int n0 = mt * 16;
  int h = threadIdx.x >> 5;
  float s = ls_p[0] * al_p[l];
  const float* D  = dist + (long)b * N_ * N_ + (long)n0 * N_;
  const u16* kvb  = KVT + (long)b * 2 * E_ * N_;
  v8f num[4] = {}, den[4] = {};
  for (int kt = 0; kt < N_ / 32; ++kt) {
    int k0 = kt * 32;
    v16bf a = load_a_exp(D + k0, nullptr, N_, s);
#pragma unroll
    for (int j = 0; j < 4; ++j) {
      int e = h * 64 + j * 16;
      num[j] = wmma_bf16(a, load_bt(kvb, N_, e, k0), num[j]);
      den[j] = wmma_bf16(a, load_bt(kvb, N_, E_ + e, k0), den[j]);
    }
  }
  int lane = threadIdx.x & 31;
  int col = lane & 15, r0 = (lane >> 4) * 8;
  long base = (long)b * N_ * E_;
#pragma unroll
  for (int j = 0; j < 4; ++j) {
    int e = h * 64 + j * 16 + col;
#pragma unroll
    for (int i = 0; i < 8; ++i) {
      int row = n0 + r0 + i;
      long o = base + (long)row * E_ + e;
      float w = num[j][i] * fast_rcp(den[j][i]);
      Y[o] = X[o] + Q[o] * w;
    }
  }
}

// ---------- instance norm over node axis ------------------------------------
// grid = B*4, block 256: 32 channels per block, 8 row-partials -> LDS reduce.
__global__ void k_inorm(const float* __restrict__ Y, const float* __restrict__ w,
                        const float* __restrict__ bb, float* __restrict__ O,
                        u16* __restrict__ OB) {
  __shared__ float s1[256], s2[256];
  int b = blockIdx.x >> 2, eg = blockIdx.x & 3;
  int tid = threadIdx.x;
  int e = eg * 32 + (tid & 31);
  int rs = tid >> 5;
  const float* Yb = Y + (long)b * N_ * E_;
  float a = 0.f, a2 = 0.f;
  for (int i = 0; i < N_ / 8; ++i) {
    float v = Yb[(long)(i * 8 + rs) * E_ + e];
    a += v; a2 += v * v;
  }
  s1[tid] = a; s2[tid] = a2;
  __syncthreads();
  for (int off = 128; off >= 32; off >>= 1) {
    if (tid < off) { s1[tid] += s1[tid + off]; s2[tid] += s2[tid + off]; }
    __syncthreads();
  }
  if (tid < 32) {
    float mean = s1[tid] * (1.f / N_);
    float var  = s2[tid] * (1.f / N_) - mean * mean;
    s1[tid] = mean;
    s2[tid] = rsqrtf(var + 1e-5f);
  }
  __syncthreads();
  float mean = s1[tid & 31], rstd = s2[tid & 31];
  float gw = w[e] * rstd, gb = bb[e];
  float* Ob = O + (long)b * N_ * E_;
  u16*  OBb = OB + (long)b * N_ * E_;
  for (int i = 0; i < N_ / 8; ++i) {
    long o = (long)(i * 8 + rs) * E_ + e;
    float v = (Yb[o] - mean) * gw + gb;
    Ob[o] = v; OBb[o] = cvt_bf16(v);
  }
}

// ---------- feed-forward GEMMs ----------------------------------------------

__global__ void k_ff1(const u16* __restrict__ HB, const u16* __restrict__ W1T,
                      const float* __restrict__ b1, u16* __restrict__ FB) {
  int b = blockIdx.x >> 5, mt = blockIdx.x & 31;
  int n0 = mt * 16;
  int wave = threadIdx.x >> 5;
  const u16* A = HB + (long)b * N_ * E_ + (long)n0 * E_;
  v8f acc[8] = {};
  for (int kt = 0; kt < E_ / 32; ++kt) {
    int k0 = kt * 32;
    v16bf a = load_a_bf16(A + k0, E_);
#pragma unroll
    for (int j = 0; j < 8; ++j)
      acc[j] = wmma_bf16(a, load_bt(W1T, E_, (wave * 8 + j) * 16, k0), acc[j]);
  }
  int lane = threadIdx.x & 31;
  int col = lane & 15, r0 = (lane >> 4) * 8;
  u16* Fb = FB + (long)b * N_ * F_;
#pragma unroll
  for (int j = 0; j < 8; ++j) {
    int c = (wave * 8 + j) * 16 + col;
    float bias = b1[c];
#pragma unroll
    for (int i = 0; i < 8; ++i) {
      float v = acc[j][i] + bias;
      v = v > 0.f ? v : 0.f;
      Fb[(long)(n0 + r0 + i) * F_ + c] = cvt_bf16(v);
    }
  }
}

__global__ void k_ff2(const u16* __restrict__ FB, const u16* __restrict__ W2T,
                      const float* __restrict__ b2, const float* __restrict__ H,
                      float* __restrict__ Y) {
  int b = blockIdx.x >> 5, mt = blockIdx.x & 31;
  int n0 = mt * 16;
  int wave = threadIdx.x >> 5;
  const u16* A = FB + (long)b * N_ * F_ + (long)n0 * F_;
  v8f acc[2] = {};
  for (int kt = 0; kt < F_ / 32; ++kt) {
    int k0 = kt * 32;
    v16bf a = load_a_bf16(A + k0, F_);
#pragma unroll
    for (int j = 0; j < 2; ++j)
      acc[j] = wmma_bf16(a, load_bt(W2T, F_, (wave * 2 + j) * 16, k0), acc[j]);
  }
  int lane = threadIdx.x & 31;
  int col = lane & 15, r0 = (lane >> 4) * 8;
  long base = (long)b * N_ * E_;
#pragma unroll
  for (int j = 0; j < 2; ++j) {
    int c = (wave * 2 + j) * 16 + col;
    float bias = b2[c];
#pragma unroll
    for (int i = 0; i < 8; ++i) {
      long o = base + (long)(n0 + r0 + i) * E_ + c;
      Y[o] = acc[j][i] + bias + H[o];
    }
  }
}

// ---------- decoder scalars --------------------------------------------------

__global__ void k_gmean(const float* __restrict__ X, float* __restrict__ G) {
  int b = blockIdx.x, e = threadIdx.x;
  const float* Xb = X + (long)b * N_ * E_;
  float s = 0.f;
  for (int n = 0; n < N_; ++n) s += Xb[(long)n * E_ + e];
  G[b * E_ + e] = s * (1.f / N_);
}

__global__ void k_gq(const float* __restrict__ G, const float* __restrict__ dWq,
                     float* __restrict__ GQ) {
  int b = blockIdx.x, e = threadIdx.x;
  float s = 0.f;
  for (int i = 0; i < E_; ++i) s += G[b * E_ + i] * dWq[i * E_ + e];
  GQ[b * E_ + e] = s;
}

// ---------- decoder AFT ------------------------------------------------------
// q[p,e] = gq[b,e] + capacity[b,p]*dWq[E,e]; aft = sigmoid(q)*num/den -> bf16 [B,P,E]
__global__ void k_aft_dec(const float* __restrict__ cdist, const float* __restrict__ mask,
                          const float* __restrict__ cap, const float* __restrict__ wlast,
                          const float* __restrict__ GQ, const u16* __restrict__ KVT,
                          u16* __restrict__ AFTB, const float* __restrict__ ls_p,
                          const float* __restrict__ al_p) {
  int b = blockIdx.x >> 5, pt = blockIdx.x & 31;
  int p0 = pt * 16;
  int h = threadIdx.x >> 5;
  float s = ls_p[0] * al_p[0];
  const float* D = cdist + (long)b * P_ * N_ + (long)p0 * N_;
  const float* M = mask  + (long)b * P_ * N_ + (long)p0 * N_;
  const u16* kvb = KVT + (long)b * 2 * E_ * N_;
  v8f num[4] = {}, den[4] = {};
  for (int kt = 0; kt < N_ / 32; ++kt) {
    int k0 = kt * 32;
    v16bf a = load_a_exp(D + k0, M + k0, N_, s);
#pragma unroll
    for (int j = 0; j < 4; ++j) {
      int e = h * 64 + j * 16;
      num[j] = wmma_bf16(a, load_bt(kvb, N_, e, k0), num[j]);
      den[j] = wmma_bf16(a, load_bt(kvb, N_, E_ + e, k0), den[j]);
    }
  }
  int lane = threadIdx.x & 31;
  int col = lane & 15, r0 = (lane >> 4) * 8;
  u16* Ab = AFTB + (long)b * P_ * E_;
#pragma unroll
  for (int j = 0; j < 4; ++j) {
    int e = h * 64 + j * 16 + col;
    float ge = GQ[b * E_ + e], we = wlast[e];
#pragma unroll
    for (int i = 0; i < 8; ++i) {
      int p = p0 + r0 + i;
      float sq = fast_rcp(1.f + __expf(-(ge + cap[b * P_ + p] * we)));
      Ab[(long)p * E_ + e] = cvt_bf16(sq * num[j][i] * fast_rcp(den[j][i]));
    }
  }
}

// ---------- score = aft @ enc^T, clip-tanh ----------------------------------
__global__ void k_score(const u16* __restrict__ AFTB, const u16* __restrict__ ENCB,
                        const float* __restrict__ cdist, const float* __restrict__ mask,
                        float* __restrict__ OUT, const float* __restrict__ ls_p,
                        const float* __restrict__ pa_p) {
  int b = blockIdx.x >> 5, pt = blockIdx.x & 31;
  int p0 = pt * 16;
  int wave = threadIdx.x >> 5;
  float spa = ls_p[0] * pa_p[0];
  const u16* A  = AFTB + (long)b * P_ * E_ + (long)p0 * E_;
  const u16* Eb = ENCB + (long)b * N_ * E_;   // enc bf16 [N,E] acts as B^T
  v8f acc[8] = {};
  for (int kt = 0; kt < E_ / 32; ++kt) {
    int k0 = kt * 32;
    v16bf a = load_a_bf16(A + k0, E_);
#pragma unroll
    for (int j = 0; j < 8; ++j)
      acc[j] = wmma_bf16(a, load_bt(Eb, E_, (wave * 8 + j) * 16, k0), acc[j]);
  }
  int lane = threadIdx.x & 31;
  int col = lane & 15, r0 = (lane >> 4) * 8;
  long base = (long)b * P_ * N_;
#pragma unroll
  for (int j = 0; j < 8; ++j) {
    int n = (wave * 8 + j) * 16 + col;
#pragma unroll
    for (int i = 0; i < 8; ++i) {
      int p = p0 + r0 + i;
      long o = base + (long)p * N_ + n;
      float sc = acc[j][i] * SQRT_E_INV + spa * cdist[o];
      OUT[o] = 10.f * tanhf(sc) + mask[o];
    }
  }
}

// ---------- row softmax over N ----------------------------------------------
__global__ void k_softmax(float* __restrict__ OUT) {
  __shared__ float red[256];
  float* R = OUT + (long)blockIdx.x * N_;
  int t = threadIdx.x;
  float x0 = R[t], x1 = R[t + 256];
  red[t] = fmaxf(x0, x1);
  __syncthreads();
  for (int off = 128; off > 0; off >>= 1) {
    if (t < off) red[t] = fmaxf(red[t], red[t + off]);
    __syncthreads();
  }
  float m = red[0];
  __syncthreads();
  float e0 = __expf(x0 - m), e1 = __expf(x1 - m);
  red[t] = e0 + e1;
  __syncthreads();
  for (int off = 128; off > 0; off >>= 1) {
    if (t < off) red[t] += red[t + off];
    __syncthreads();
  }
  float inv = fast_rcp(red[0]);
  R[t] = e0 * inv; R[t + 256] = e1 * inv;
}

// ---------- host launch ------------------------------------------------------

extern "C" void kernel_launch(void* const* d_in, const int* in_sizes, int n_in,
                              void* d_out, int out_size, void* d_ws, size_t ws_size,
                              hipStream_t stream) {
  (void)in_sizes; (void)n_in; (void)out_size; (void)ws_size;
  const float* data      = (const float*)d_in[0];
  const float* dist      = (const float*)d_in[1];
  const float* cdist     = (const float*)d_in[2];
  const float* cap       = (const float*)d_in[3];
  const float* mask      = (const float*)d_in[4];
  const float* log_scale = (const float*)d_in[5];
  const float* emb_W     = (const float*)d_in[6];
  const float* emb_b     = (const float*)d_in[7];
  const float* Wq        = (const float*)d_in[8];
  const float* Wk        = (const float*)d_in[9];
  const float* Wv        = (const float*)d_in[10];
  const float* aft_alpha = (const float*)d_in[11];
  const float* n1_w      = (const float*)d_in[12];
  const float* n1_b      = (const float*)d_in[13];
  const float* ff_W1     = (const float*)d_in[14];
  const float* ff_b1     = (const float*)d_in[15];
  const float* ff_W2     = (const float*)d_in[16];
  const float* ff_b2     = (const float*)d_in[17];
  const float* n2_w      = (const float*)d_in[18];
  const float* n2_b      = (const float*)d_in[19];
  const float* dWq       = (const float*)d_in[20];
  const float* dWk       = (const float*)d_in[21];
  const float* dWv       = (const float*)d_in[22];
  const float* dec_alpha = (const float*)d_in[23];
  const float* prob_a    = (const float*)d_in[24];
  float* OUT = (float*)d_out;

  char* ws = (char*)d_ws;
  size_t off = 0;
  auto alloc = [&](size_t bytes) {
    void* p = ws + off;
    off = (off + bytes + 255) & ~(size_t)255;
    return p;
  };
  float* X   = (float*)alloc((size_t)B_ * N_ * E_ * 4);
  u16*   XB  = (u16*)  alloc((size_t)B_ * N_ * E_ * 2);
  float* Q   = (float*)alloc((size_t)B_ * N_ * E_ * 4);
  u16*   KVT = (u16*)  alloc((size_t)B_ * 2 * E_ * N_ * 2);
  float* Y   = (float*)alloc((size_t)B_ * N_ * E_ * 4);
  float* H   = (float*)alloc((size_t)B_ * N_ * E_ * 4);
  u16*   HB  = (u16*)  alloc((size_t)B_ * N_ * E_ * 2);
  u16*   FB  = (u16*)  alloc((size_t)B_ * N_ * F_ * 2);
  float* GM  = (float*)alloc((size_t)B_ * E_ * 4);
  float* GQb = (float*)alloc((size_t)B_ * E_ * 4);
  u16*  AFTB = (u16*)  alloc((size_t)B_ * P_ * E_ * 2);
  u16*   WQT = (u16*)  alloc((size_t)L_ * E_ * E_ * 2);
  u16*   WKT = (u16*)  alloc((size_t)L_ * E_ * E_ * 2);
  u16*   WVT = (u16*)  alloc((size_t)L_ * E_ * E_ * 2);
  u16*   W1T = (u16*)  alloc((size_t)L_ * E_ * F_ * 2);
  u16*   W2T = (u16*)  alloc((size_t)L_ * F_ * E_ * 2);
  u16*   DKT = (u16*)  alloc((size_t)E_ * E_ * 2);
  u16*   DVT = (u16*)  alloc((size_t)E_ * E_ * 2);

  auto cdiv = [](long a, long b) { return (int)((a + b - 1) / b); };

  // one-time weight convert+transpose to bf16
  k_cvt_t<<<cdiv((long)L_ * E_ * E_, 256), 256, 0, stream>>>(Wq, WQT, E_, E_, L_);
  k_cvt_t<<<cdiv((long)L_ * E_ * E_, 256), 256, 0, stream>>>(Wk, WKT, E_, E_, L_);
  k_cvt_t<<<cdiv((long)L_ * E_ * E_, 256), 256, 0, stream>>>(Wv, WVT, E_, E_, L_);
  k_cvt_t<<<cdiv((long)L_ * E_ * F_, 256), 256, 0, stream>>>(ff_W1, W1T, E_, F_, L_);
  k_cvt_t<<<cdiv((long)L_ * F_ * E_, 256), 256, 0, stream>>>(ff_W2, W2T, F_, E_, L_);
  k_cvt_t<<<cdiv((long)E_ * E_, 256), 256, 0, stream>>>(dWk, DKT, E_, E_, 1);
  k_cvt_t<<<cdiv((long)E_ * E_, 256), 256, 0, stream>>>(dWv, DVT, E_, E_, 1);

  k_embed<<<(B_ * N_ * E_) / 256, 256, 0, stream>>>(data, emb_W, emb_b, X, XB);

  for (int l = 0; l < L_; ++l) {
    k_qkv<true><<<B_ * 32, 128, 0, stream>>>(XB, WQT + (size_t)l * E_ * E_,
                                             WKT + (size_t)l * E_ * E_,
                                             WVT + (size_t)l * E_ * E_, Q, KVT);
    k_aft_enc<<<B_ * 32, 64, 0, stream>>>(dist, X, Q, KVT, Y, log_scale, aft_alpha, l);
    k_inorm<<<B_ * 4, 256, 0, stream>>>(Y, n1_w + l * E_, n1_b + l * E_, H, HB);
    k_ff1<<<B_ * 32, 128, 0, stream>>>(HB, W1T + (size_t)l * E_ * F_, ff_b1 + l * F_, FB);
    k_ff2<<<B_ * 32, 128, 0, stream>>>(FB, W2T + (size_t)l * F_ * E_, ff_b2 + l * E_, H, Y);
    k_inorm<<<B_ * 4, 256, 0, stream>>>(Y, n2_w + l * E_, n2_b + l * E_, X, XB);
  }

  // decoder
  k_qkv<false><<<B_ * 32, 128, 0, stream>>>(XB, nullptr, DKT, DVT, nullptr, KVT);
  k_gmean<<<B_, E_, 0, stream>>>(X, GM);
  k_gq<<<B_, E_, 0, stream>>>(GM, dWq, GQb);
  k_aft_dec<<<B_ * 32, 64, 0, stream>>>(cdist, mask, cap, dWq + (size_t)E_ * E_,
                                        GQb, KVT, AFTB, log_scale, dec_alpha);
  k_score<<<B_ * 32, 128, 0, stream>>>(AFTB, XB, cdist, mask, OUT, log_scale, prob_a);
  k_softmax<<<B_ * P_, 256, 0, stream>>>(OUT);
}
